// KolmogorovArnoldLayer_85633057948371
// MI455X (gfx1250) — matrix operand order repaired and checked by reading, not compiled, not run on hardware
//
#include <hip/hip_runtime.h>

// ---------------------------------------------------------------------------
// KAN layer for MI455X (gfx1250, wave32, WMMA):
//   out = silu(x) @ wb + spline(x) @ ws
// fused as one GEMM  C[4096,512] = A_cat[4096,512] @ B_cat[512,512]
// split-bf16 (hi/lo) precision: C = Ah*Bh + Ah*Bl + Al*Bh  (f32 accum)
// GEMM uses double-buffered LDS fed by CDNA5 async global->LDS copies
// (GLOBAL_LOAD_ASYNC_TO_LDS_B128, tracked by ASYNCcnt).
// ---------------------------------------------------------------------------

typedef __attribute__((ext_vector_type(16))) __bf16 v16bf;
typedef __attribute__((ext_vector_type(8)))  float  v8f;
typedef __attribute__((ext_vector_type(4)))  int    v4i;

union Frag { v16bf v; v4i q[2]; };

static constexpr int BATCH = 4096;
static constexpr int IN    = 256;
static constexpr int OUT   = 512;   // N
static constexpr int KDIM  = 512;   // concatenated K = 2*IN
static constexpr int NKNOT = 64;
static constexpr int NBAS  = 60;    // NKNOT - DEGREE - 1

static constexpr int TM = 128;      // macro tile M
static constexpr int TN = 128;      // macro tile N
static constexpr int TK = 32;       // k-step == WMMA K
static constexpr int NT = KDIM / TK;
static constexpr int APITCH = 40;   // LDS row pitch in bf16 (32 + 8 pad, 80B)

// builtin expects v4i addrspace(1)* src, v4i addrspace(3)* dst
typedef __attribute__((address_space(1))) v4i gv4i;
typedef __attribute__((address_space(3))) v4i lv4i;

// ---------------------------------------------------------------------------
// helpers
// ---------------------------------------------------------------------------
__device__ __forceinline__ void store_split(__bf16* ph, __bf16* pl, float v) {
    __bf16 h = (__bf16)v;
    *ph = h;
    *pl = (__bf16)(v - (float)h);
}

__device__ __forceinline__ float knotv(const float* kn, int idx, float h) {
    if (idx < 0)  return -1.0f + (float)idx * h;
    if (idx > 63) return  1.0f + (float)(idx - 63) * h;
    return kn[idx];
}

__device__ __forceinline__ v8f wmma_bf16(v16bf a, v16bf b, v8f c) {
    return __builtin_amdgcn_wmma_f32_16x16x32_bf16(false, a, false, b,
                                                   (short)0, c, false, false);
}

// async 16B global -> LDS copy (per-lane addresses), tracked by ASYNCcnt
__device__ __forceinline__ void async16(const __bf16* g, __bf16* l) {
#if __has_builtin(__builtin_amdgcn_global_load_async_to_lds_b128)
    __builtin_amdgcn_global_load_async_to_lds_b128((gv4i*)g, (lv4i*)l, 0, 0);
#else
    unsigned lofs = (unsigned)(size_t)l;   // low 32 bits of generic = LDS addr
    asm volatile("global_load_async_to_lds_b128 %0, %1, off"
                 :: "v"(lofs), "v"((unsigned long long)(size_t)g)
                 : "memory");
#endif
}

__device__ __forceinline__ void wait_async0() {
#if __has_builtin(__builtin_amdgcn_s_wait_asynccnt)
    __builtin_amdgcn_s_wait_asynccnt(0);
#else
    asm volatile("s_wait_asynccnt 0x0" ::: "memory");
#endif
}

// ---------------------------------------------------------------------------
// Kernel 1: per-element activation — silu(x) and cubic B-spline value
// ---------------------------------------------------------------------------
__global__ __launch_bounds__(256)
void kan_act(const float* __restrict__ x, const float* __restrict__ cps,
             const float* __restrict__ knots,
             __bf16* __restrict__ Ah, __bf16* __restrict__ Al) {
    int idx = blockIdx.x * 256 + threadIdx.x;       // over BATCH*IN
    if (idx >= BATCH * IN) return;
    int b = idx >> 8;
    int i = idx & (IN - 1);
    float xv = x[idx];

    float silu = xv / (1.0f + expf(-xv));

    const float h = 2.0f / 63.0f;
    int j = (int)floorf((xv + 1.0f) / h);
    j = min(max(j, 0), 62);    // knot span: t_j <= x < t_{j+1}

    float Nv[4];
    Nv[0] = 1.0f;
#pragma unroll
    for (int d = 1; d <= 3; ++d) {
        float saved = 0.0f;
#pragma unroll
        for (int r = 0; r < 3; ++r) {
            if (r < d) {
                float tl = knotv(knots, j + r + 1, h);
                float tr = knotv(knots, j + r + 1 - d, h);
                float temp = Nv[r] / (tl - tr);
                Nv[r] = saved + (tl - xv) * temp;
                saved = (xv - tr) * temp;
            }
        }
        Nv[d] = saved;
    }
    float sp = 0.0f;
#pragma unroll
    for (int r = 0; r < 4; ++r) {
        int k = j - 3 + r;
        if (k >= 0 && k < NBAS) sp += cps[i * NKNOT + k] * Nv[r];
    }

    int off = b * KDIM + i;
    store_split(Ah + off,      Al + off,      silu);
    store_split(Ah + off + IN, Al + off + IN, sp);
}

// ---------------------------------------------------------------------------
// Kernel 2: build B_cat^T [OUT][KDIM] as bf16 hi/lo
// ---------------------------------------------------------------------------
__global__ __launch_bounds__(256)
void kan_prepB(const float* __restrict__ wb, const float* __restrict__ wsm,
               __bf16* __restrict__ Bth, __bf16* __restrict__ Btl) {
    int idx = blockIdx.x * 256 + threadIdx.x;       // over KDIM*OUT
    int k = idx >> 9;
    int n = idx & (OUT - 1);
    float v = (k < IN) ? wb[k * OUT + n] : wsm[(k - IN) * OUT + n];
    __bf16 h = (__bf16)v;
    Bth[n * KDIM + k] = h;
    Btl[n * KDIM + k] = (__bf16)(v - (float)h);
}

// ---------------------------------------------------------------------------
// Kernel 3: split-bf16 WMMA GEMM, double-buffered async LDS pipeline
// 128x128 macro tile, 8 waves (4M x 2N), wave tile 32x64 (2x4 fragments).
// ---------------------------------------------------------------------------
struct __align__(16) LdsBuf {
    __bf16 Ah[TM * APITCH];
    __bf16 Al[TM * APITCH];
    __bf16 Bh[TN * APITCH];
    __bf16 Bl[TN * APITCH];
};

__device__ __forceinline__ v16bf load_a_frag(const __bf16* base, int m0, int lane) {
    int m = m0 + (lane & 15);
    int half = lane >> 4;
    const __bf16* p = base + m * APITCH + half * 8;
    Frag f;
    f.q[0] = *(const v4i*)(p);
    f.q[1] = *(const v4i*)(p + 16);
    return f.v;
}

__device__ __forceinline__ v16bf load_b_frag(const __bf16* base, int n0, int lane) {
    int n = n0 + (lane & 15);
    int half = lane >> 4;
    const __bf16* p = base + n * APITCH + half * 8;
    Frag f;
    f.q[0] = *(const v4i*)(p);
    f.q[1] = *(const v4i*)(p + 16);
    return f.v;
}

__global__ __launch_bounds__(256)
void kan_gemm(const __bf16* __restrict__ Ah, const __bf16* __restrict__ Al,
              const __bf16* __restrict__ Bth, const __bf16* __restrict__ Btl,
              float* __restrict__ C) {
    __shared__ LdsBuf lds[2];

    const int tid  = threadIdx.x;
    const int lane = tid & 31;
    const int wave = tid >> 5;
    const int mw   = wave >> 1;    // 0..3 -> 32-row slice
    const int nw   = wave & 1;     // 0..1 -> 64-col slice

    const int m0 = blockIdx.x * TM;    // grid.x = 32
    const int n0 = blockIdx.y * TN;    // grid.y = 4

    // per-thread staging offsets (2 x 16B chunks per array per stage)
    int lo[2], gA[2], gB[2];
#pragma unroll
    for (int p = 0; p < 2; ++p) {
        int c   = tid + p * 256;     // 0..511
        int row = c >> 2;            // 0..127
        int kc  = c & 3;             // 0..3 (8 bf16 each)
        lo[p] = row * APITCH + kc * 8;
        gA[p] = (m0 + row) * KDIM + kc * 8;
        gB[p] = (n0 + row) * KDIM + kc * 8;
    }

    auto stage = [&](int kt, int buf) {
#pragma unroll
        for (int p = 0; p < 2; ++p) {
            int ka = gA[p] + kt * TK;
            int kb = gB[p] + kt * TK;
            async16(Ah  + ka, lds[buf].Ah + lo[p]);
            async16(Al  + ka, lds[buf].Al + lo[p]);
            async16(Bth + kb, lds[buf].Bh + lo[p]);
            async16(Btl + kb, lds[buf].Bl + lo[p]);
        }
    };

    v8f acc[2][4] = {};

    stage(0, 0);
    for (int kt = 0; kt < NT; ++kt) {
        const int cur = kt & 1;
        wait_async0();            // this wave's stage-kt copies landed in LDS
        __syncthreads();          // everyone's copies landed; prev reads done
        if (kt + 1 < NT) stage(kt + 1, cur ^ 1);

        const LdsBuf& L = lds[cur];
        v16bf ah[2], al[2], bh[4], bl[4];
#pragma unroll
        for (int mf = 0; mf < 2; ++mf) {
            ah[mf] = load_a_frag(L.Ah, mw * 32 + mf * 16, lane);
            al[mf] = load_a_frag(L.Al, mw * 32 + mf * 16, lane);
        }
#pragma unroll
        for (int nf = 0; nf < 4; ++nf) {
            bh[nf] = load_b_frag(L.Bh, nw * 64 + nf * 16, lane);
            bl[nf] = load_b_frag(L.Bl, nw * 64 + nf * 16, lane);
        }

#pragma unroll
        for (int mf = 0; mf < 2; ++mf) {
#pragma unroll
            for (int nf = 0; nf < 4; ++nf) {
                acc[mf][nf] = wmma_bf16(ah[mf], bh[nf], acc[mf][nf]);
                acc[mf][nf] = wmma_bf16(ah[mf], bl[nf], acc[mf][nf]);
                acc[mf][nf] = wmma_bf16(al[mf], bh[nf], acc[mf][nf]);
            }
        }
    }

    // store C: lane r -> M = r + 8*half, N = lane%16 (32-bit C/D layout)
    const int half = lane >> 4;
    const int lm   = lane & 15;
#pragma unroll
    for (int mf = 0; mf < 2; ++mf) {
#pragma unroll
        for (int nf = 0; nf < 4; ++nf) {
            int col   = n0 + nw * 64 + nf * 16 + lm;
            int rbase = m0 + mw * 32 + mf * 16 + half * 8;
#pragma unroll
            for (int r = 0; r < 8; ++r)
                C[(rbase + r) * OUT + col] = acc[mf][nf][r];
        }
    }
}

// ---------------------------------------------------------------------------
// launch
// ---------------------------------------------------------------------------
extern "C" void kernel_launch(void* const* d_in, const int* in_sizes, int n_in,
                              void* d_out, int out_size, void* d_ws, size_t ws_size,
                              hipStream_t stream) {
    const float* x     = (const float*)d_in[0];
    const float* wb    = (const float*)d_in[1];
    const float* wsm   = (const float*)d_in[2];
    const float* cps   = (const float*)d_in[3];
    const float* knots = (const float*)d_in[4];
    float* out = (float*)d_out;

    __bf16* Ah  = (__bf16*)d_ws;
    __bf16* Al  = Ah  + (size_t)BATCH * KDIM;
    __bf16* Bth = Al  + (size_t)BATCH * KDIM;
    __bf16* Btl = Bth + (size_t)OUT * KDIM;

    kan_act<<<(BATCH * IN) / 256, 256, 0, stream>>>(x, cps, knots, Ah, Al);
    kan_prepB<<<(KDIM * OUT) / 256, 256, 0, stream>>>(wb, wsm, Bth, Btl);

    dim3 grid(BATCH / TM, OUT / TN);   // 32 x 4
    kan_gemm<<<grid, 256, 0, stream>>>(Ah, Al, Bth, Btl, out);
}
